// Capsule_28097676051143
// MI455X (gfx1250) — matrix operand order repaired and checked by reading, not compiled, not run on hardware
//
#include <hip/hip_runtime.h>

// Problem constants (from reference)
#define B_ 64
#define I_ 512
#define E_ 256
#define N_ 32
#define D_ 64
#define M_ (B_ * I_)    // 32768 GEMM rows  (b,i)
#define NC_ (N_ * D_)   // 2048  GEMM cols  (n,d)

typedef __bf16 v16bf_t __attribute__((ext_vector_type(16)));
typedef __bf16 v8bf_t  __attribute__((ext_vector_type(8)));
typedef __bf16 v4bf_t  __attribute__((ext_vector_type(4)));
typedef float  v8f_t   __attribute__((ext_vector_type(8)));

union V16 { v16bf_t v; v8bf_t h[2]; };

// ---------------------------------------------------------------------------
// Kernel 1: u_hat[b,n,i,d] = sum_e u[b,i,e] * W[e, n*D+d]   (bf16 out, f32 acc)
// One GEMM: A=[32768,256] fp32, B=[256,2048] fp32, C stored bf16 in capsule
// layout. 128x128 block tile, 8 waves, each wave = 2x4 of 16x16 WMMA tiles.
// ---------------------------------------------------------------------------
__global__ __launch_bounds__(256) void caps_uhat_gemm(
    const float* __restrict__ U, const float* __restrict__ W,
    __bf16* __restrict__ uhat)
{
    __shared__ __align__(16) __bf16 Ash[128][40];   // [row][k]  (+8 pad)
    __shared__ __align__(16) __bf16 Bsh[128][40];   // [col][k]  transposed

    const int t       = threadIdx.x;
    const int rowBase = blockIdx.y * 128;           // along (b,i)
    const int colBase = blockIdx.x * 128;           // along (n,d)

    const int lane = t & 31;
    const int wid  = t >> 5;
    const int wy   = wid >> 1;       // 0..3 : 32-row slab
    const int wx   = wid & 1;        // 0..1 : 64-col slab
    const int mrow = lane & 15;      // A-row / B-col / C-col within 16x16
    const int kh   = lane >> 4;      // K-half selector

    v8f_t acc[2][4] = {};

    for (int ks = 0; ks < E_ / 32; ++ks) {
        const int k0 = ks * 32;

        // stage A: 128 rows x 32 k, fp32 -> bf16 (float4 per thread, x4)
        #pragma unroll
        for (int l = 0; l < 4; ++l) {
            int idx = t + l * 256;              // 0..1023
            int r   = idx >> 3;                 // 0..127
            int kg  = (idx & 7) << 2;           // 0,4,..,28
            const float4 v = *(const float4*)&U[(size_t)(rowBase + r) * E_ + k0 + kg];
            Ash[r][kg + 0] = (__bf16)v.x;
            Ash[r][kg + 1] = (__bf16)v.y;
            Ash[r][kg + 2] = (__bf16)v.z;
            Ash[r][kg + 3] = (__bf16)v.w;
        }
        // stage B transposed: Bsh[col][k] from W[k][col].
        // One column + 4 consecutive k per thread: coalesced b32 global loads
        // (consecutive lanes -> consecutive cols), one ds_store_b64 per group.
        #pragma unroll
        for (int l = 0; l < 4; ++l) {
            int idx = t + l * 256;              // 0..1023
            int col = idx & 127;
            int kg  = (idx >> 7) << 2;          // 0,4,..,28
            const float* wp = &W[(size_t)(k0 + kg) * NC_ + colBase + col];
            v4bf_t pk;
            pk[0] = (__bf16)wp[0 * NC_];
            pk[1] = (__bf16)wp[1 * NC_];
            pk[2] = (__bf16)wp[2 * NC_];
            pk[3] = (__bf16)wp[3 * NC_];
            *(v4bf_t*)&Bsh[col][kg] = pk;       // 8B aligned (stride 80B)
        }
        // L2 prefetch of next K-slab (gfx1250 global_prefetch_b8)
        if (ks + 1 < E_ / 32) {
            __builtin_prefetch(&U[(size_t)(rowBase + (t >> 1)) * E_ + k0 + 32], 0, 1);
            __builtin_prefetch(&W[(size_t)(k0 + 32 + (t >> 3)) * NC_ + colBase + ((t & 7) << 4)], 0, 1);
        }
        __syncthreads();

        // fragments per CDNA5 16-bit layouts
        V16 fa[2], fb[4];
        #pragma unroll
        for (int mt = 0; mt < 2; ++mt) {
            int row = wy * 32 + mt * 16 + mrow;
            fa[mt].h[0] = *(const v8bf_t*)&Ash[row][kh * 8];        // K = kh*8 .. +7
            fa[mt].h[1] = *(const v8bf_t*)&Ash[row][16 + kh * 8];   // K = 16+kh*8 ..
        }
        #pragma unroll
        for (int nt = 0; nt < 4; ++nt) {
            int col = wx * 64 + nt * 16 + mrow;
            fb[nt].h[0] = *(const v8bf_t*)&Bsh[col][kh * 8];        // K = kh*8 .. +7
            fb[nt].h[1] = *(const v8bf_t*)&Bsh[col][16 + kh * 8];   // K = 16+kh*8 ..
        }
        #pragma unroll
        for (int mt = 0; mt < 2; ++mt)
            #pragma unroll
            for (int nt = 0; nt < 4; ++nt)
                acc[mt][nt] = __builtin_amdgcn_wmma_f32_16x16x32_bf16(
                    false, fa[mt].v, false, fb[nt].v,
                    (short)0, acc[mt][nt], false, false);
        __syncthreads();
    }

    // store C into u_hat[b][n][i][d] as bf16
    const int mb = (lane >> 4) * 8;   // C: lanes 16-31 hold M = 8 + r
    #pragma unroll
    for (int mt = 0; mt < 2; ++mt) {
        #pragma unroll
        for (int nt = 0; nt < 4; ++nt) {
            int colG = colBase + wx * 64 + nt * 16 + mrow;
            int ncap = colG >> 6;
            int d    = colG & 63;
            #pragma unroll
            for (int r = 0; r < 8; ++r) {
                int rowG = rowBase + wy * 32 + mt * 16 + mb + r;
                int bb   = rowG >> 9;        // / I_
                int ii   = rowG & 511;       // % I_
                size_t idx = (((size_t)bb * N_ + ncap) * I_ + ii) * D_ + d;
                uhat[idx] = (__bf16)acc[mt][nt][r];
            }
        }
    }
}

// ---------------------------------------------------------------------------
// Kernel 2: dynamic routing (3 iters) + squash. One block per batch element.
// c[32][512] (bf16) and o[32][64] (f32) live in LDS; u_hat reads hit L2.
// ---------------------------------------------------------------------------
__global__ __launch_bounds__(512) void caps_route(
    const __bf16* __restrict__ uhat, float* __restrict__ out)
{
    __shared__ __align__(16) __bf16 c_sh[N_][I_];   // 32 KB
    __shared__ __align__(16) float o_sh[N_][D_];    // 8 KB
    __shared__ float scale_sh[N_];

    const int t = threadIdx.x;
    const int b = blockIdx.x;
    const __bf16* ub = uhat + (size_t)b * N_ * I_ * D_;

    // iter 0 coupling: softmax(0) over n  ->  uniform 1/N
    for (int idx = t; idx < N_ * I_; idx += 512)
        ((__bf16*)c_sh)[idx] = (__bf16)(1.0f / (float)N_);
    __syncthreads();

    // o-pass mapping: one capsule n and 4 consecutive d per thread
    const int n_o = t >> 4;            // 0..31
    const int dg  = (t & 15) << 2;     // 0,4,..,60

    for (int it = 0; it < 3; ++it) {
        // ---- o[n,dg..dg+3] = sum_i c[n,i] * u_hat[n,i,dg..dg+3] ----
        // u_hat comes in as 8B v4bf loads (16 lanes x 8B = one 128B row),
        // c comes in as one 8B LDS load per 4 i's.
        {
            const __bf16* p = ub + (size_t)n_o * I_ * D_ + dg;
            float a0 = 0.f, a1 = 0.f, a2 = 0.f, a3 = 0.f;
            for (int i = 0; i < I_; i += 4) {
                v4bf_t c4 = *(const v4bf_t*)&c_sh[n_o][i];
                #pragma unroll
                for (int q = 0; q < 4; ++q) {
                    v4bf_t u4 = *(const v4bf_t*)(p + (size_t)(i + q) * D_);
                    float cf = (float)c4[q];
                    a0 += cf * (float)u4[0];
                    a1 += cf * (float)u4[1];
                    a2 += cf * (float)u4[2];
                    a3 += cf * (float)u4[3];
                }
            }
            __syncthreads();   // previous-iteration reads of o_sh are done
            *(float4*)&o_sh[n_o][dg] = make_float4(a0, a1, a2, a3);
        }
        __syncthreads();

        if (it == 2) break;

        // ---- L2-normalize o per capsule ----
        if (t < N_) {
            float s = 0.f;
            #pragma unroll
            for (int dd = 0; dd < D_; ++dd) s += o_sh[t][dd] * o_sh[t][dd];
            float inv = rsqrtf(fmaxf(s, 1e-12f));
            #pragma unroll
            for (int dd = 0; dd < D_; ++dd) o_sh[t][dd] *= inv;
        }
        __syncthreads();

        // ---- logits b[n,i] = <o[n,:], u_hat[n,i,:]>, softmax over n ----
        {
            const int i = t;               // 512 threads == I_
            float lg[N_];
            float mx = -3.4e38f;
            #pragma unroll
            for (int n = 0; n < N_; ++n) {
                const __bf16* p = ub + ((size_t)n * I_ + i) * D_;
                float s = 0.f;
                #pragma unroll
                for (int dd = 0; dd < D_; dd += 8) {
                    v8bf_t v = *(const v8bf_t*)(p + dd);   // 16B aligned
                    #pragma unroll
                    for (int e = 0; e < 8; ++e)
                        s += o_sh[n][dd + e] * (float)v[e];
                }
                lg[n] = s;
                mx = fmaxf(mx, s);
            }
            float sum = 0.f;
            #pragma unroll
            for (int n = 0; n < N_; ++n) { float e = __expf(lg[n] - mx); lg[n] = e; sum += e; }
            float inv = 1.f / sum;
            #pragma unroll
            for (int n = 0; n < N_; ++n) c_sh[n][i] = (__bf16)(lg[n] * inv);
        }
        __syncthreads();
    }

    // ---- squash(o) and write out[b,n,d] ----
    if (t < N_) {
        float s2 = 0.f;
        #pragma unroll
        for (int dd = 0; dd < D_; ++dd) s2 += o_sh[t][dd] * o_sh[t][dd];
        scale_sh[t] = s2 / ((1.f + s2) * sqrtf(s2));
    }
    __syncthreads();
    for (int idx = t; idx < N_ * D_; idx += 512) {
        int n = idx >> 6;
        out[(size_t)b * N_ * D_ + idx] = scale_sh[n] * o_sh[n][idx & 63];
    }
}

// ---------------------------------------------------------------------------
extern "C" void kernel_launch(void* const* d_in, const int* in_sizes, int n_in,
                              void* d_out, int out_size, void* d_ws, size_t ws_size,
                              hipStream_t stream) {
    (void)in_sizes; (void)n_in; (void)out_size; (void)ws_size;
    const float* U = (const float*)d_in[0];   // [64,512,256] f32
    const float* W = (const float*)d_in[1];   // [1,256,2048] f32
    __bf16* uhat   = (__bf16*)d_ws;           // [64,32,512,64] bf16 = 134 MB

    dim3 gemmGrid(NC_ / 128, M_ / 128);       // 16 x 256 blocks
    caps_uhat_gemm<<<gemmGrid, 256, 0, stream>>>(U, W, uhat);
    caps_route<<<B_, 512, 0, stream>>>(uhat, (float*)d_out);
}
